// SGNP_62405874811519
// MI455X (gfx1250) — matrix operand order repaired and checked by reading, not compiled, not run on hardware
//
#include <hip/hip_runtime.h>
#include <hip/hip_bf16.h>

// ---------------------------------------------------------------------------
// Problem constants (match reference)
// ---------------------------------------------------------------------------
#define BATCH   2
#define N_CTX   4096
#define N_TST   1024
#define D_X     64
#define D_S     3
#define KSEL    32

#define TQ      16          // query tile (WMMA M)
#define TN      16          // ctx tile  (WMMA N)
#define NTHREADS 256        // 8 waves (wave32)
#define NWAVES  (NTHREADS / 32)

typedef __attribute__((ext_vector_type(16))) _Float16 v16h;
typedef __attribute__((ext_vector_type(8)))  float    v8f;

// Dynamic LDS layout (bytes):
//   dsq16 : TQ*N_CTX u16  = 131072   (f16 bit patterns; u16-sortable since >=0)
//   rnx   : N_CTX f32     =  16384
//   qnx/qsv/qtv           :    320
//   sel   : TQ*KSEL int   =   2048
// total 149824 B  ->  2 workgroups per 320KB WGP (occupancy 2)
#define SMEM_BYTES (TQ*N_CTX*2 + (N_CTX + TQ + TQ*3 + TQ)*4 + TQ*KSEL*4)

// pack two float4 (8 contiguous f32) into v16h elements [base, base+8)
__device__ __forceinline__ void pack8(v16h& dst, int base, float4 x, float4 y) {
    dst[base + 0] = (_Float16)x.x; dst[base + 1] = (_Float16)x.y;
    dst[base + 2] = (_Float16)x.z; dst[base + 3] = (_Float16)x.w;
    dst[base + 4] = (_Float16)y.x; dst[base + 5] = (_Float16)y.y;
    dst[base + 6] = (_Float16)y.z; dst[base + 7] = (_Float16)y.w;
}

__global__ void __launch_bounds__(NTHREADS)
knn_edges_kernel(const float* __restrict__ qx,   // (B, nq, 64)
                 const float* __restrict__ qs,   // (B, nq, 3)
                 const float* __restrict__ qt,   // (B, nq)
                 const float* __restrict__ rx,   // (B, N_CTX, 64)
                 const float* __restrict__ rs,   // (B, N_CTX, 3)
                 const float* __restrict__ rt,   // (B, N_CTX)
                 const unsigned char* __restrict__ mask_ctx, // (B, N_CTX)
                 int nq, int edge_base, int recv_base,
                 int*   __restrict__ out_send,
                 int*   __restrict__ out_recv,
                 float* __restrict__ out_dx,
                 float* __restrict__ out_ds,
                 float* __restrict__ out_dt,
                 unsigned char* __restrict__ out_mask)
{
    extern __shared__ unsigned char smem_raw[];
    unsigned short* dsq16 = (unsigned short*)smem_raw;            // [TQ][N_CTX]
    float* rnx = (float*)(smem_raw + (size_t)TQ * N_CTX * 2);     // [N_CTX]
    float* qnx = rnx + N_CTX;                                     // [TQ]
    float* qsv = qnx + TQ;                                        // [TQ][3]
    float* qtv = qsv + TQ * 3;                                    // [TQ]
    int*   sel = (int*)(qtv + TQ);                                // [TQ][KSEL]

    const float INF = __builtin_inff();

    const int tiles_per_batch = nq / TQ;
    const int b     = blockIdx.x / tiles_per_batch;
    const int qtile = blockIdx.x % tiles_per_batch;

    const float* qxb = qx + (size_t)b * nq * D_X;
    const float* qsb = qs + (size_t)b * nq * D_S;
    const float* qtb = qt + (size_t)b * nq;
    const float* rxb = rx + (size_t)b * N_CTX * D_X;
    const float* rsb = rs + (size_t)b * N_CTX * D_S;
    const float* rtb = rt + (size_t)b * N_CTX;
    const unsigned char* mkb = mask_ctx + (size_t)b * N_CTX;

    const int tid  = threadIdx.x;
    const int wave = tid >> 5;
    const int lane = tid & 31;

    // ---- stage per-query data (norms, s, t) -------------------------------
    if (tid < TQ) {
        const int q = qtile * TQ + tid;
        const float4* r4 = (const float4*)(qxb + (size_t)q * D_X);
        float acc = 0.f;
        #pragma unroll
        for (int d4 = 0; d4 < D_X / 4; ++d4) {
            const float4 v = r4[d4];
            acc += v.x * v.x + v.y * v.y + v.z * v.z + v.w * v.w;
        }
        qnx[tid] = acc;
        qsv[tid * 3 + 0] = qsb[q * D_S + 0];
        qsv[tid * 3 + 1] = qsb[q * D_S + 1];
        qsv[tid * 3 + 2] = qsb[q * D_S + 2];
        qtv[tid] = qtb[q];
    }
    // ---- ctx x-norms, cooperative -----------------------------------------
    for (int n = tid; n < N_CTX; n += NTHREADS) {
        const float4* r4 = (const float4*)(rxb + (size_t)n * D_X);
        float acc = 0.f;
        #pragma unroll
        for (int d4 = 0; d4 < D_X / 4; ++d4) {
            const float4 v = r4[d4];
            acc += v.x * v.x + v.y * v.y + v.z * v.z + v.w * v.w;
        }
        rnx[n] = acc;
    }
    __syncthreads();

    // ---- WMMA phase: D = Q(16x64) x R^T(64x16) per ctx tile ---------------
    // 16-bit A-fragment (16x32), ISA 7.12.2: lane m = lane&15; lanes>=16 hold
    // the K=+8 half; elements [0..7] -> K=kb..kb+7, [8..15] -> K=kb+16..kb+23.
    const int m      = lane & 15;
    const int kbaseA = (lane < 16) ? 0 : 8;
    const int kbaseB = (lane < 16) ? 0 : 8;
    const int qrow   = qtile * TQ + m;

    const float4* qv4 = (const float4*)(qxb + (size_t)qrow * D_X);
    v16h a0, a1;  // K = 0..31 and 32..63 of this lane's query row
    {
        const int i0 = kbaseA >> 2;          // 0 or 2
        pack8(a0, 0, qv4[i0],      qv4[i0 + 1]);     // K = kb .. kb+7
        pack8(a0, 8, qv4[i0 + 4],  qv4[i0 + 5]);     // K = kb+16 .. kb+23
        pack8(a1, 0, qv4[i0 + 8],  qv4[i0 + 9]);     // +32
        pack8(a1, 8, qv4[i0 + 12], qv4[i0 + 13]);
    }

    const int ntiles = N_CTX / TN;     // 256
    for (int j = wave; j < ntiles; j += NWAVES) {
        const int n = j * TN + m;      // this lane's ctx column (B/C share n)

        if (j + NWAVES < ntiles)       // global_prefetch for next tile's row
            __builtin_prefetch(&rxb[(size_t)(j + NWAVES) * TN * D_X + m * D_X], 0, 3);

        // B-fragment (32x16): same K order as A (elements [0..7] -> K=kb..kb+7, [8..15] -> K=kb+16..kb+23)
        const float4* rv4 = (const float4*)(rxb + (size_t)n * D_X);
        v16h b0, b1;
        {
            const int i0 = kbaseB >> 2;      // 0 or 2
            pack8(b0, 0, rv4[i0],      rv4[i0 + 1]);
            pack8(b0, 8, rv4[i0 + 4],  rv4[i0 + 5]);
            pack8(b1, 0, rv4[i0 + 8],  rv4[i0 + 9]);
            pack8(b1, 8, rv4[i0 + 12], rv4[i0 + 13]);
        }

        v8f c = {};
        c = __builtin_amdgcn_wmma_f32_16x16x32_f16(false, a0, false, b0,
                                                   (short)0, c, false, false);
        c = __builtin_amdgcn_wmma_f32_16x16x32_f16(false, a1, false, b1,
                                                   (short)0, c, false, false);

        // epilogue: combine into d_sq, store as f16 bits (u16-sortable, >=0)
        const float rn  = rnx[n];
        const float rs0 = rsb[n * D_S + 0];
        const float rs1 = rsb[n * D_S + 1];
        const float rs2 = rsb[n * D_S + 2];
        const float rtv = rtb[n];
        const bool  mk  = mkb[n] != 0;
        const int   moff = (lane < 16) ? 0 : 8;

        #pragma unroll
        for (int vg = 0; vg < 8; ++vg) {
            const int   mm  = vg + moff;      // C layout: VGPR vg -> row
            const float dot = c[vg];
            float dx = qnx[mm] + rn - 2.0f * dot;
            const float e0 = qsv[mm * 3 + 0] - rs0;
            const float e1 = qsv[mm * 3 + 1] - rs1;
            const float e2 = qsv[mm * 3 + 2] - rs2;
            float ds_ = e0 * e0 + e1 * e1 + e2 * e2;
            float dt_ = rtv - qtv[mm];
            if (!mk) { dx = INF; ds_ = INF; dt_ = INF; }
            if (dt_ > 0.0f) dt_ = INF;
            const float dsqv = dx * dx + ds_ * ds_ + dt_ * dt_;
            const _Float16 hv = (_Float16)dsqv;   // overflow->inf only for far pts
            dsq16[mm * N_CTX + n] = __builtin_bit_cast(unsigned short, hv);
        }
    }
    __syncthreads();

    // ---- exact top-K, wave-private, integer compares on f16 bits ----------
    // Wave w owns rows w and w+8.  Lane l scans uint2 word-pairs l, l+32, ...
    // (ds_load_b64, all 64 banks covered -> conflict-free; 4 candidates/load).
    // After each extraction only the owning lane rescans its strip.
    for (int mrow = wave; mrow < TQ; mrow += NWAVES) {
        uint2* row2 = (uint2*)(dsq16 + (size_t)mrow * N_CTX);   // 1024 uint2
        unsigned bv; int bi;
        auto scan = [&]() {
            bv = 0xFFFFFFFFu; bi = 0x7fffffff;
            for (int i = 0; i < N_CTX / 4 / 32; ++i) {          // 32 iters
                const int   w  = i * 32 + lane;
                const uint2 pv = row2[w];
                const int   c4 = w * 4;
                const unsigned v0 = pv.x & 0xFFFFu, v1 = pv.x >> 16;
                const unsigned v2 = pv.y & 0xFFFFu, v3 = pv.y >> 16;
                if (v0 < bv) { bv = v0; bi = c4 + 0; }
                if (v1 < bv) { bv = v1; bi = c4 + 1; }
                if (v2 < bv) { bv = v2; bi = c4 + 2; }
                if (v3 < bv) { bv = v3; bi = c4 + 3; }
            }
        };
        scan();
        for (int kk = 0; kk < KSEL; ++kk) {
            unsigned rv = bv; int ri = bi;
            #pragma unroll
            for (int off = 16; off > 0; off >>= 1) {
                const unsigned ov = __shfl_xor(rv, off, 32);
                const int      oi = __shfl_xor(ri, off, 32);
                if (ov < rv || (ov == rv && oi < ri)) { rv = ov; ri = oi; }
            }
            if (lane == 0) sel[mrow * KSEL + kk] = ri;
            if (ri == bi) {                  // unique owner (disjoint strips)
                uint2 v = row2[ri >> 2];     // knock out element: set 0xFFFF
                switch (ri & 3) {
                    case 0: v.x = v.x | 0x0000FFFFu; v.x |= 0xFFFFu;
                            v.x = (v.x & 0xFFFF0000u) | 0xFFFFu; break;
                    case 1: v.x = (v.x & 0x0000FFFFu) | 0xFFFF0000u; break;
                    case 2: v.y = (v.y & 0xFFFF0000u) | 0xFFFFu;     break;
                    default: v.y = (v.y & 0x0000FFFFu) | 0xFFFF0000u; break;
                }
                row2[ri >> 2] = v;
                scan();                      // refresh this lane's local min
            }
        }
    }
    __syncthreads();

    // ---- emit edges: recompute exact f32 distances for 512 selected pairs -
    for (int p = tid; p < TQ * KSEL; p += NTHREADS) {
        const int mrow = p >> 5;
        const int kk   = p & (KSEL - 1);
        const int n    = sel[p];
        const int q    = qtile * TQ + mrow;

        const float4* q4 = (const float4*)(qxb + (size_t)q * D_X);
        const float4* r4 = (const float4*)(rxb + (size_t)n * D_X);
        float accx = 0.f;
        #pragma unroll
        for (int d4 = 0; d4 < D_X / 4; ++d4) {
            const float4 a = q4[d4], c = r4[d4];
            const float x0 = a.x - c.x, x1 = a.y - c.y;
            const float x2 = a.z - c.z, x3 = a.w - c.w;
            accx += x0 * x0 + x1 * x1 + x2 * x2 + x3 * x3;
        }
        float accs = 0.f;
        #pragma unroll
        for (int d = 0; d < D_S; ++d) {
            const float df = qsb[q * D_S + d] - rsb[n * D_S + d];
            accs += df * df;
        }
        const float dtv = rtb[n] - qtb[q];
        const bool  mk  = mkb[n] != 0;

        float d_x = mk ? accx : INF;
        float d_s = mk ? accs : INF;
        float d_t = mk ? dtv  : INF;
        if (d_t > 0.0f) d_t = INF;
        const bool em = mk && (dtv <= 0.0f);

        const int e = edge_base + (b * nq + q) * KSEL + kk;
        out_send[e] = n + b * N_CTX;
        out_recv[e] = recv_base + b * nq + q;
        out_dx[e]   = d_x;
        out_ds[e]   = d_s;
        out_dt[e]   = d_t;
        out_mask[e] = em ? 1 : 0;
    }
}

// ---------------------------------------------------------------------------
// Launch
// ---------------------------------------------------------------------------
extern "C" void kernel_launch(void* const* d_in, const int* in_sizes, int n_in,
                              void* d_out, int out_size, void* d_ws, size_t ws_size,
                              hipStream_t stream)
{
    const float*         x_ctx  = (const float*)d_in[0];
    const float*         s_ctx  = (const float*)d_in[1];
    const float*         t_ctx  = (const float*)d_in[2];
    const unsigned char* m_ctx  = (const unsigned char*)d_in[3];
    const float*         x_test = (const float*)d_in[4];
    const float*         s_test = (const float*)d_in[5];
    const float*         t_test = (const float*)d_in[6];

    const int E_cc = BATCH * N_CTX * KSEL;   // 262144
    const int E_ct = BATCH * N_TST * KSEL;   //  65536
    const int E    = E_cc + E_ct;            // 327680

    // Output layout (concatenated, reference return order):
    // senders(i32) | receivers(i32) | d_x(f32) | d_s(f32) | d_t(f32) | mask(u8)
    int*   out_send = (int*)d_out;
    int*   out_recv = out_send + E;
    float* out_dx   = (float*)d_out + 2 * (size_t)E;
    float* out_ds   = (float*)d_out + 3 * (size_t)E;
    float* out_dt   = (float*)d_out + 4 * (size_t)E;
    unsigned char* out_mask = (unsigned char*)((float*)d_out + 5 * (size_t)E);

    // Pass 1: ctx -> ctx
    {
        dim3 grid(BATCH * (N_CTX / TQ));     // 512 workgroups
        knn_edges_kernel<<<grid, NTHREADS, SMEM_BYTES, stream>>>(
            x_ctx, s_ctx, t_ctx, x_ctx, s_ctx, t_ctx, m_ctx,
            N_CTX, /*edge_base=*/0, /*recv_base=*/0,
            out_send, out_recv, out_dx, out_ds, out_dt, out_mask);
    }
    // Pass 2: test -> ctx
    {
        dim3 grid(BATCH * (N_TST / TQ));     // 128 workgroups
        knn_edges_kernel<<<grid, NTHREADS, SMEM_BYTES, stream>>>(
            x_test, s_test, t_test, x_ctx, s_ctx, t_ctx, m_ctx,
            N_TST, /*edge_base=*/E_cc, /*recv_base=*/BATCH * N_CTX,
            out_send, out_recv, out_dx, out_ds, out_dt, out_mask);
    }
}
